// MultiHeadAttention_84293028152083
// MI455X (gfx1250) — matrix operand order repaired and checked by reading, not compiled, not run on hardware
//
#include <hip/hip_runtime.h>
#include <hip/hip_bf16.h>

// ---------------- problem constants ----------------
#define B_  2
#define S_  4096
#define E_  1024
#define H_  16
#define DK_ 64
#define BS_ 128
#define NB_ 32            // S/BS
#define QT_ 16            // query tile per workgroup
#define HD_ 1024          // H*DK

// ---------------- WMMA vector types ----------------
typedef __bf16 v16bf __attribute__((ext_vector_type(16)));
typedef float  v8f   __attribute__((ext_vector_type(8)));
typedef float  v2f   __attribute__((ext_vector_type(2)));
typedef int    v4i_  __attribute__((vector_size(4 * sizeof(int))));

#if __has_builtin(__builtin_amdgcn_wmma_f32_16x16x4_f32)
#define HAVE_WMMA4 1
#else
#define HAVE_WMMA4 0
#endif

#if __has_builtin(__builtin_amdgcn_global_load_async_to_lds_b128) && \
    __has_builtin(__builtin_amdgcn_s_wait_asynccnt)
#define HAVE_ASYNC 1
#else
#define HAVE_ASYNC 0
#endif

#define GLOBAL_AS __attribute__((address_space(1)))
#define LDS_AS    __attribute__((address_space(3)))

union FragU { uint4 q[2]; v16bf v; };

// A-matrix fragment (16x32 bf16), blocked layout:
// lane = (row = lane&15, half = lane>>4); pass p = rowbase + half*16.
// VGPR0..7 <- K[half*16 .. +15] (16 contiguous elements)
__device__ __forceinline__ v16bf afrag_from(const __hip_bfloat16* p) {
  FragU f;
  f.q[0] = *(const uint4*)(p);
  f.q[1] = *(const uint4*)(p + 8);
  return f.v;
}

// B-matrix fragment (32x16 bf16), blocked layout:
// lane = (col = lane&15, half = lane>>4); pass p = colbase + half*16.
// VGPR0..7 <- K[half*16 .. +15] (16 contiguous elements)
__device__ __forceinline__ v16bf bfrag_from(const __hip_bfloat16* p) {
  FragU f;
  const uint4* u = (const uint4*)p;
  f.q[0] = u[0];
  f.q[1] = u[1];
  return f.v;
}

// ---------------- fp32 -> bf16 convert ----------------
__global__ void cvt_f32_to_bf16(const float* __restrict__ in,
                                __hip_bfloat16* __restrict__ out, int n) {
  int i = (blockIdx.x * blockDim.x + threadIdx.x) * 4;
  if (i + 3 < n) {
    float4 f = *(const float4*)(in + i);
    out[i + 0] = __float2bfloat16(f.x);
    out[i + 1] = __float2bfloat16(f.y);
    out[i + 2] = __float2bfloat16(f.z);
    out[i + 3] = __float2bfloat16(f.w);
  } else {
    for (int k = i; k < n; ++k) out[k] = __float2bfloat16(in[k]);
  }
}

// ---------------- bf16 WMMA GEMM: C[M,N] = A[M,K] * W[N,K]^T ----------------
// 128x128 tile, 256 threads = 8 waves in 4x2 (M x N) grid; each wave 32x64.
// Double-buffered LDS staging via global_load_async_to_lds_b128 (ASYNCcnt),
// falling back to synchronous staging + global_prefetch_b8.
// mode 0: store bf16 [b,h,s,d]             (Q projection)
// mode 1: store bf16 [b,h,s,d] + [b,h,d,s] (K projection + transposed copy)
// mode 2: store fp32 row-major [M,N]       (output projection)
__global__ __launch_bounds__(256) void gemm_bf16_wmma(
    const __hip_bfloat16* __restrict__ A,
    const __hip_bfloat16* __restrict__ Wt,
    int M, int N, int K, int mode,
    __hip_bfloat16* __restrict__ outb,
    __hip_bfloat16* __restrict__ outb2,
    float* __restrict__ outf) {
#if HAVE_ASYNC
  __shared__ __align__(16) __hip_bfloat16 sA[2 * 128 * 32];
  __shared__ __align__(16) __hip_bfloat16 sB[2 * 128 * 32];
#else
  __shared__ __align__(16) __hip_bfloat16 sA[128 * 32];
  __shared__ __align__(16) __hip_bfloat16 sB[128 * 32];
#endif

  const int tn = N >> 7;
  const int tile_n = blockIdx.x % tn;
  const int tile_m = blockIdx.x / tn;
  const int m0 = tile_m << 7, n0 = tile_n << 7;
  const int tid = threadIdx.x, lane = tid & 31, w = tid >> 5;
  const int wm = w >> 1, wn = w & 1;
  const int half = lane >> 4, lc = lane & 15;
  const int srow = tid >> 1, sseg = (tid & 1) << 4;  // 2 threads/row, 16 elems each

  v8f acc[2][4] = {};

  const __hip_bfloat16* gA = A  + (size_t)(m0 + srow) * K + sseg;
  const __hip_bfloat16* gB = Wt + (size_t)(n0 + srow) * K + sseg;

#if HAVE_ASYNC
  // Each thread owns one 32-byte row segment per matrix -> 4 async b128 ops
  // per stage per wave-lane; async loads complete in order.
  auto issue_stage = [&](int buf, int k0) {
    const __hip_bfloat16* a = gA + k0;
    const __hip_bfloat16* b = gB + k0;
    __hip_bfloat16* la = sA + buf * (128 * 32) + srow * 32 + sseg;
    __hip_bfloat16* lb = sB + buf * (128 * 32) + srow * 32 + sseg;
    __builtin_amdgcn_global_load_async_to_lds_b128(
        (GLOBAL_AS v4i_*)(a), (LDS_AS v4i_*)(la), 0, 0);
    __builtin_amdgcn_global_load_async_to_lds_b128(
        (GLOBAL_AS v4i_*)(a + 8), (LDS_AS v4i_*)(la + 8), 0, 0);
    __builtin_amdgcn_global_load_async_to_lds_b128(
        (GLOBAL_AS v4i_*)(b), (LDS_AS v4i_*)(lb), 0, 0);
    __builtin_amdgcn_global_load_async_to_lds_b128(
        (GLOBAL_AS v4i_*)(b + 8), (LDS_AS v4i_*)(lb + 8), 0, 0);
  };

  int p = 0;
  issue_stage(0, 0);
  for (int k0 = 0; k0 < K; k0 += 32) {
    if (k0 + 32 < K) {
      issue_stage(p ^ 1, k0 + 32);           // prefetch next stage
      __builtin_amdgcn_s_wait_asynccnt(4);   // oldest 4 (stage p) landed
    } else {
      __builtin_amdgcn_s_wait_asynccnt(0);
    }
    __syncthreads();                         // stage p visible to all waves

    const __hip_bfloat16* bA = sA + p * (128 * 32);
    const __hip_bfloat16* bB = sB + p * (128 * 32);
    v16bf af[2], bfr[4];
#pragma unroll
    for (int i = 0; i < 2; ++i)
      af[i] = afrag_from(bA + (wm * 32 + i * 16 + lc) * 32 + half * 16);
#pragma unroll
    for (int j = 0; j < 4; ++j)
      bfr[j] = bfrag_from(bB + (wn * 64 + j * 16 + lc) * 32 + half * 16);
#pragma unroll
    for (int i = 0; i < 2; ++i)
#pragma unroll
      for (int j = 0; j < 4; ++j)
        acc[i][j] = __builtin_amdgcn_wmma_f32_16x16x32_bf16(
            false, af[i], false, bfr[j], (short)0, acc[i][j], false, false);
    __syncthreads();                         // reads done before p is reused
    p ^= 1;
  }
#else
  for (int k0 = 0; k0 < K; k0 += 32) {
    uint4 a0 = ((const uint4*)(gA + k0))[0];
    uint4 a1 = ((const uint4*)(gA + k0))[1];
    uint4 b0 = ((const uint4*)(gB + k0))[0];
    uint4 b1 = ((const uint4*)(gB + k0))[1];
    if (k0 + 32 < K) {
      __builtin_prefetch(gA + k0 + 32, 0, 1);
      __builtin_prefetch(gB + k0 + 32, 0, 1);
    }
    __syncthreads();
    *(uint4*)(sA + srow * 32 + sseg)     = a0;
    *(uint4*)(sA + srow * 32 + sseg + 8) = a1;
    *(uint4*)(sB + srow * 32 + sseg)     = b0;
    *(uint4*)(sB + srow * 32 + sseg + 8) = b1;
    __syncthreads();

    v16bf af[2], bfr[4];
#pragma unroll
    for (int i = 0; i < 2; ++i)
      af[i] = afrag_from(sA + (wm * 32 + i * 16 + lc) * 32 + half * 16);
#pragma unroll
    for (int j = 0; j < 4; ++j)
      bfr[j] = bfrag_from(sB + (wn * 64 + j * 16 + lc) * 32 + half * 16);
#pragma unroll
    for (int i = 0; i < 2; ++i)
#pragma unroll
      for (int j = 0; j < 4; ++j)
        acc[i][j] = __builtin_amdgcn_wmma_f32_16x16x32_bf16(
            false, af[i], false, bfr[j], (short)0, acc[i][j], false, false);
  }
#endif

  // Epilogue. C-tile layout: lane -> (row = 8*half + v, col = lane&15)
#pragma unroll
  for (int i = 0; i < 2; ++i)
#pragma unroll
    for (int j = 0; j < 4; ++j)
#pragma unroll
      for (int v = 0; v < 8; ++v) {
        int m   = m0 + wm * 32 + i * 16 + half * 8 + v;
        int col = n0 + wn * 64 + j * 16 + lc;
        float val = acc[i][j][v];
        if (mode == 2) {
          outf[(size_t)m * N + col] = val;
        } else {
          int bb = m >> 12, ss = m & (S_ - 1);
          int hh = col >> 6, dd = col & 63;
          __hip_bfloat16 bv = __float2bfloat16(val);
          outb[(((size_t)bb * H_ + hh) * S_ + ss) * DK_ + dd] = bv;
          if (mode == 1)
            outb2[(((size_t)bb * H_ + hh) * DK_ + dd) * S_ + ss] = bv;
        }
      }
}

// ---------------- blockwise talking-heads attention ----------------
// grid: B * NB * (BS/QT) = 512 blocks; 512 threads = 16 waves, wave w = head h.
// Wave h: 16 queries x 384-key window. QK^T and PV via v_wmma_f32_16x16x32_bf16.
// Talking-heads mixers (Tb pre-softmax, Ta post-softmax) via
// v_wmma_f32_16x16x4_f32: mixed = T(16x16) @ S(16 heads x elements),
// chunked through LDS (128-key chunks). Wave h owns the q=h row of col-tiles
// during mixing; a single LDS scatter redistributes rows back to head-waves.
__global__ __launch_bounds__(512) void attn_blockwise(
    const __hip_bfloat16* __restrict__ Qb,   // [B,H,S,DK]
    const __hip_bfloat16* __restrict__ Kb,   // [B,H,S,DK]
    const __hip_bfloat16* __restrict__ KTb,  // [B,H,DK,S]
    const unsigned char* __restrict__ qmask, // [B,S] (bool)
    const float* __restrict__ Tbm, const float* __restrict__ Tam,
    const float* __restrict__ relp, const float* __restrict__ absp,
    __hip_bfloat16* __restrict__ attn_out)   // [B,S,H*DK]
{
  extern __shared__ char smem[];
  float*          arena  = (float*)smem;                      // [16][16][128] f32
  float*          arena2 = (float*)(smem + 131072);           // [16][16][128] f32
  __hip_bfloat16* pM     = (__hip_bfloat16*)(smem + 131072);  // [16][16][128] bf16 (aliases arena2)
  float*          sTb    = (float*)(smem + 262144);
  float*          sTa    = (float*)(smem + 262144 + 1024);

  const int tid  = threadIdx.x;
  const int lane = tid & 31;
  const int h    = tid >> 5;           // wave == head
  const int half = lane >> 4;
  const int lc   = lane & 15;

  if (tid < 256) { sTb[tid] = Tbm[tid]; sTa[tid] = Tam[tid]; }

  const int bx    = blockIdx.x;
  const int qt    = bx & 7;
  const int n     = (bx >> 3) & 31;
  const int b     = bx >> 8;
  const int qbase = n * BS_ + qt * QT_;
  const int kw0   = n * BS_ - BS_;     // global kpos of window index 0

  const float rel_a = relp[0];
  const float abs_a = absp[0];

  __syncthreads();

#if HAVE_WMMA4
  // A-fragments of Tb/Ta for the 16x16x4 f32 mixer.
  // 32-bit A 16x4 layout: lane row = lc; VGPR0 = K(2*half), VGPR1 = K(2*half+1).
  v2f tbf[4], taf[4];
#pragma unroll
  for (int s = 0; s < 4; ++s) {
    int g = s * 4 + 2 * half;
    tbf[s].x = sTb[lc * 16 + g];
    tbf[s].y = sTb[lc * 16 + g + 1];
    taf[s].x = sTa[lc * 16 + g];
    taf[s].y = sTa[lc * 16 + g + 1];
  }
#endif

  // Q A-fragments (contraction dim = DK = 64 -> two 16x32 fragments)
  const __hip_bfloat16* qrow =
      Qb + (((size_t)b * H_ + h) * S_ + (qbase + lc)) * DK_;
  v16bf qf[2];
#pragma unroll
  for (int s2 = 0; s2 < 2; ++s2)
    qf[s2] = afrag_from(qrow + s2 * 32 + half * 16);

  const __hip_bfloat16* Kh = Kb + (((size_t)b * H_ + h) * S_) * DK_;

  float sreg[24][8];  // mixed scores, full 384-key row in registers

  // ---- Phase 1: raw scores -> decay/bias -> Tb mix (WMMA f32) -> mask ----
#pragma unroll
  for (int c = 0; c < 3; ++c) {       // 128-key chunks
#pragma unroll
    for (int tt = 0; tt < 8; ++tt) {  // 16-key tiles
      int kr  = kw0 + (c * 8 + tt) * 16 + lc;
      int krc = kr < 0 ? 0 : (kr >= S_ ? S_ - 1 : kr);  // clamp; masked later
      const __hip_bfloat16* krow = Kh + (size_t)krc * DK_;
      v8f cacc = {};
#pragma unroll
      for (int s2 = 0; s2 < 2; ++s2) {
        v16bf bf = bfrag_from(krow + s2 * 32 + half * 16);
        cacc = __builtin_amdgcn_wmma_f32_16x16x32_bf16(
            false, qf[s2], false, bf, (short)0, cacc, false, false);
      }
#pragma unroll
      for (int v = 0; v < 8; ++v) {
        int qpos = qbase + half * 8 + v;
        int kpos = kw0 + (c * 8 + tt) * 16 + lc;
        int dd = qpos - kpos; dd = dd < 0 ? -dd : dd;
        float s = (cacc[v] * 0.125f) *
                  __builtin_amdgcn_rcpf(1.0f + rel_a * (float)dd);
        if (kpos < qpos) s -= abs_a;  // absolute bias (pre-mix, head-indep)
        arena[(h * 16 + half * 8 + v) * 128 + tt * 16 + lc] = s;
      }
    }
    __syncthreads();
#if HAVE_WMMA4
    // Tb mix: wave h mixes the q=h row, all 8 key tiles, all 16 output heads.
#pragma unroll
    for (int kt = 0; kt < 8; ++kt) {
      v8f d = {};
#pragma unroll
      for (int s = 0; s < 4; ++s) {
        int g = s * 4 + 2 * half;
        v2f bfr;
        bfr.x = arena[((g + 0) * 16 + h) * 128 + kt * 16 + lc];
        bfr.y = arena[((g + 1) * 16 + h) * 128 + kt * 16 + lc];
        d = __builtin_amdgcn_wmma_f32_16x16x4_f32(
            false, tbf[s], false, bfr, (short)0, d, false, false);
      }
      // D rows = output heads; scatter back to [head][q][k] layout
#pragma unroll
      for (int v = 0; v < 8; ++v)
        arena2[((half * 8 + v) * 16 + h) * 128 + kt * 16 + lc] = d[v];
    }
    __syncthreads();
    // redistribute: wave h reads its own head's mixed row + mask
#pragma unroll
    for (int tt = 0; tt < 8; ++tt)
#pragma unroll
      for (int v = 0; v < 8; ++v) {
        float acc = arena2[(h * 16 + half * 8 + v) * 128 + tt * 16 + lc];
        int qpos = qbase + half * 8 + v;
        int kpos = kw0 + c * 128 + tt * 16 + lc;
        bool valid = (kpos <= qpos) && (kpos >= 0) && (kpos < S_) &&
                     (qmask[(size_t)b * S_ + qpos] != 0);
        sreg[c * 8 + tt][v] = valid ? acc : -1000.0f;
      }
#else
    // VALU fallback mixer
#pragma unroll
    for (int tt = 0; tt < 8; ++tt)
#pragma unroll
      for (int v = 0; v < 8; ++v) {
        int row = half * 8 + v, col = tt * 16 + lc;
        float acc = 0.f;
#pragma unroll
        for (int g = 0; g < 16; ++g)
          acc = fmaf(sTb[h * 16 + g], arena[(g * 16 + row) * 128 + col], acc);
        int qpos = qbase + row;
        int kpos = kw0 + c * 128 + col;
        bool valid = (kpos <= qpos) && (kpos >= 0) && (kpos < S_) &&
                     (qmask[(size_t)b * S_ + qpos] != 0);
        sreg[c * 8 + tt][v] = valid ? acc : -1000.0f;
      }
#endif
    __syncthreads();
  }

  // ---- Phase 2: softmax over 384 keys (in registers; wave32 shfl) ----
#pragma unroll
  for (int v = 0; v < 8; ++v) {
    float m = -3.4e38f;
#pragma unroll
    for (int t = 0; t < 24; ++t) m = fmaxf(m, sreg[t][v]);
#pragma unroll
    for (int d = 8; d >= 1; d >>= 1) m = fmaxf(m, __shfl_xor(m, d, 32));
    float sum = 0.f;
#pragma unroll
    for (int t = 0; t < 24; ++t) {
      float e = __expf(sreg[t][v] - m);
      sreg[t][v] = e;
      sum += e;
    }
#pragma unroll
    for (int d = 8; d >= 1; d >>= 1) sum += __shfl_xor(sum, d, 32);
    float r = 1.0f / sum;
#pragma unroll
    for (int t = 0; t < 24; ++t) sreg[t][v] *= r;
  }

  // ---- Phase 3: Ta mix (WMMA f32) + P@V (WMMA bf16) ----
  v8f o[4] = {};
  const __hip_bfloat16* KTh = KTb + (((size_t)b * H_ + h) * DK_) * S_;
#pragma unroll
  for (int c = 0; c < 3; ++c) {
    // stage p (f32) in [head][q][k] layout
#pragma unroll
    for (int tt = 0; tt < 8; ++tt)
#pragma unroll
      for (int v = 0; v < 8; ++v)
        arena[(h * 16 + half * 8 + v) * 128 + tt * 16 + lc] =
            sreg[c * 8 + tt][v];
    __syncthreads();
#if HAVE_WMMA4
#pragma unroll
    for (int kt = 0; kt < 8; ++kt) {
      v8f d = {};
#pragma unroll
      for (int s = 0; s < 4; ++s) {
        int g = s * 4 + 2 * half;
        v2f bfr;
        bfr.x = arena[((g + 0) * 16 + h) * 128 + kt * 16 + lc];
        bfr.y = arena[((g + 1) * 16 + h) * 128 + kt * 16 + lc];
        d = __builtin_amdgcn_wmma_f32_16x16x4_f32(
            false, taf[s], false, bfr, (short)0, d, false, false);
      }
      // write mixed probs straight into PV A-staging [head][q][k] bf16
#pragma unroll
      for (int v = 0; v < 8; ++v)
        pM[((half * 8 + v) * 16 + h) * 128 + kt * 16 + lc] =
            __float2bfloat16(d[v]);
    }
#else
#pragma unroll
    for (int tt = 0; tt < 8; ++tt)
#pragma unroll
      for (int v = 0; v < 8; ++v) {
        int row = half * 8 + v, col = tt * 16 + lc;
        float acc = 0.f;
#pragma unroll
        for (int g = 0; g < 16; ++g)
          acc = fmaf(sTa[h * 16 + g], arena[(g * 16 + row) * 128 + col], acc);
        pM[(h * 16 + row) * 128 + col] = __float2bfloat16(acc);
      }
#endif
    __syncthreads();
    // PV: A = mixed probs (LDS), B = K^T (global, contiguous along keys)
#pragma unroll
    for (int ks = 0; ks < 4; ++ks) {  // 32-key contraction slices
      v16bf af = afrag_from(pM + (h * 16 + lc) * 128 + ks * 32 + half * 16);
#pragma unroll
      for (int j = 0; j < 4; ++j) {   // 16-wide d tiles
        int k0  = kw0 + c * 128 + ks * 32 + half * 16;
        int k0c = k0 < 0 ? 0 : (k0 > S_ - 16 ? S_ - 16 : k0);  // p==0 there
        const __hip_bfloat16* krow = KTh + (size_t)(j * 16 + lc) * S_;
        v16bf bf = bfrag_from(krow + k0c);
        o[j] = __builtin_amdgcn_wmma_f32_16x16x32_bf16(
            false, af, false, bf, (short)0, o[j], false, false);
      }
    }
    __syncthreads();
  }

  // ---- store attention output as [B, S, H*DK] bf16 (ready for Wo GEMM) ----
#pragma unroll
  for (int j = 0; j < 4; ++j)
#pragma unroll
    for (int v = 0; v < 8; ++v) {
      int q   = qbase + half * 8 + v;
      int col = h * DK_ + j * 16 + lc;
      attn_out[((size_t)b * S_ + q) * HD_ + col] = __float2bfloat16(o[j][v]);
    }
}

// ---------------- launch ----------------
extern "C" void kernel_launch(void* const* d_in, const int* in_sizes, int n_in,
                              void* d_out, int out_size, void* d_ws, size_t ws_size,
                              hipStream_t stream) {
  const float*         x     = (const float*)d_in[0];
  const unsigned char* qmask = (const unsigned char*)d_in[1];  // bool array
  const float*         Wq    = (const float*)d_in[2];
  const float*         Wk    = (const float*)d_in[3];
  const float*         Wo    = (const float*)d_in[4];
  const float*         Tb    = (const float*)d_in[5];
  const float*         Ta    = (const float*)d_in[6];
  const float*         rel_a = (const float*)d_in[7];
  const float*         abs_a = (const float*)d_in[8];

  // workspace layout (~86 MiB total)
  char* ws = (char*)d_ws;
  size_t off = 0;
  auto take = [&](size_t bytes) -> char* {
    char* p = ws + off;
    off += (bytes + 255) & ~(size_t)255;
    return p;
  };
  const int NX = B_ * S_ * E_;       // 8388608
  const int NW = E_ * HD_;           // 1048576
  __hip_bfloat16* x_bf    = (__hip_bfloat16*)take((size_t)NX * 2);
  __hip_bfloat16* Wq_bf   = (__hip_bfloat16*)take((size_t)NW * 2);
  __hip_bfloat16* Wk_bf   = (__hip_bfloat16*)take((size_t)NW * 2);
  __hip_bfloat16* Wo_bf   = (__hip_bfloat16*)take((size_t)NW * 2);
  __hip_bfloat16* Q_bf    = (__hip_bfloat16*)take((size_t)NX * 2);
  __hip_bfloat16* K_bf    = (__hip_bfloat16*)take((size_t)NX * 2);
  __hip_bfloat16* KT_bf   = (__hip_bfloat16*)take((size_t)NX * 2);
  __hip_bfloat16* attn_bf = (__hip_bfloat16*)take((size_t)NX * 2);

  // 1) convert fp32 inputs to bf16
  cvt_f32_to_bf16<<<NX / 1024, 256, 0, stream>>>(x, x_bf, NX);
  cvt_f32_to_bf16<<<NW / 1024, 256, 0, stream>>>(Wq, Wq_bf, NW);
  cvt_f32_to_bf16<<<NW / 1024, 256, 0, stream>>>(Wk, Wk_bf, NW);
  cvt_f32_to_bf16<<<NW / 1024, 256, 0, stream>>>(Wo, Wo_bf, NW);

  const int M = B_ * S_;                           // 8192
  const int gemm_blocks = (M / 128) * (E_ / 128);  // 512

  // 2) Q = x @ Wq^T  -> [B,H,S,DK] bf16
  gemm_bf16_wmma<<<gemm_blocks, 256, 0, stream>>>(
      x_bf, Wq_bf, M, HD_, E_, 0, Q_bf, nullptr, nullptr);
  // 3) K = x @ Wk^T  -> [B,H,S,DK] + transposed [B,H,DK,S]
  gemm_bf16_wmma<<<gemm_blocks, 256, 0, stream>>>(
      x_bf, Wk_bf, M, HD_, E_, 1, K_bf, KT_bf, nullptr);

  // 4) blockwise talking-heads attention
  const int attn_blocks = B_ * NB_ * (BS_ / QT_);  // 512
  const size_t attn_lds = 131072 + 131072 + 2048;  // arena + arena2/pM + Tb + Ta
  attn_blockwise<<<attn_blocks, 512, attn_lds, stream>>>(
      Q_bf, K_bf, KT_bf, qmask, Tb, Ta, rel_a, abs_a, attn_bf);

  // 5) out = attn @ Wo^T -> fp32 [B,S,E]
  gemm_bf16_wmma<<<gemm_blocks, 256, 0, stream>>>(
      attn_bf, Wo_bf, M, E_, HD_, 2, nullptr, nullptr, (float*)d_out);
}
